// Model_6743098655445
// MI455X (gfx1250) — matrix operand order repaired and checked by reading, not compiled, not run on hardware
//
#include <hip/hip_runtime.h>
#include <hip/hip_bf16.h>
#include <math.h>

typedef __attribute__((ext_vector_type(16))) __bf16 v16bf;
typedef __attribute__((ext_vector_type(8)))  __bf16 v8bf;
typedef __attribute__((ext_vector_type(4)))  __bf16 v4bf;
typedef __attribute__((ext_vector_type(8)))  float  v8f;
typedef __attribute__((ext_vector_type(4)))  float  v4f;

#define THREADS 256

// Problem constants
constexpr int Bq  = 32;    // Qlen
constexpr int D   = 128;   // hidden
constexpr int LV  = 384;   // L*V keys
constexpr int H   = 4;     // heads
// LDS pitches (elements), padded to avoid bank conflicts
constexpr int KV_P = 136;  // bf16 pitch for 128-wide rows
constexpr int KH_P = 40;   // bf16 pitch for K_h (384 x 32)
constexpr int VT_P = 392;  // bf16 pitch for V_h^T (32 x 384)
constexpr int S_P  = 392;  // f32 pitch for scores (32 x 384)
constexpr int O_P  = 136;  // f32 pitch for O (32 x 128)

// LDS byte offsets (all 16B aligned)
constexpr size_t OFF_KV  = 0;
constexpr size_t OFF_Q0  = OFF_KV + (size_t)LV * KV_P * 2;   // raw q (bf16)
constexpr size_t OFF_QP  = OFF_Q0 + (size_t)Bq * KV_P * 2;   // projected Q (bf16)
constexpr size_t OFF_KH  = OFF_QP + (size_t)Bq * KV_P * 2;   // K_h (bf16)
constexpr size_t OFF_VT  = OFF_KH + (size_t)LV * KH_P * 2;   // V_h^T (bf16)
constexpr size_t OFF_O   = OFF_VT + (size_t)32 * VT_P * 2;   // O accum (f32)
constexpr size_t OFF_SC  = OFF_O  + (size_t)Bq * O_P * 4;    // scratch: weights / scores
constexpr size_t SCR_SZ  = 50176;                            // max(wq 34816, wk+wv 17408, sS 50176)
constexpr size_t OFF_RED = OFF_SC + SCR_SZ;                  // softmax max/inv-sum
constexpr size_t SMEM_BYTES = OFF_RED + 256;

__device__ __forceinline__ v16bf cat16(v8bf lo, v8bf hi) {
  return __builtin_shufflevector(lo, hi, 0,1,2,3,4,5,6,7,8,9,10,11,12,13,14,15);
}

// A-operand (16x32 bf16), tile base = row 0 / col 0, per ISA wave32 layout:
// lane<16: M=lane, K runs [0..7],[16..23]; lane>=16: M=lane-16, K runs [8..15],[24..31]
__device__ __forceinline__ v16bf lds_load_A(const __bf16* t, int pitch, int lane) {
  const int m  = lane & 15;
  const int g8 = (lane & 16) >> 1;           // 0 or 8
  const __bf16* p = t + m * pitch + g8;
  v8bf lo = *(const v8bf*)p;                 // K = 8g .. 8g+7
  v8bf hi = *(const v8bf*)(p + 16);          // K = 16+8g .. 23+8g
  return cat16(lo, hi);
}

// B-operand (32x16 bf16) from a transposed (N x K row-major) LDS tile:
// lane<16: K runs [0..7],[16..23]; lane>=16: K runs [8..15],[24..31]; N = lane&15
__device__ __forceinline__ v16bf lds_load_BT(const __bf16* tT, int pitch, int lane) {
  const int n   = lane & 15;
  const int g8  = (lane & 16) >> 1;          // 0 or 8
  const __bf16* p = tT + n * pitch + g8;
  v8bf lo = *(const v8bf*)p;
  v8bf hi = *(const v8bf*)(p + 16);
  return cat16(lo, hi);
}

// A-operand from an f32 LDS tile, converted to bf16 in-register
__device__ __forceinline__ v16bf lds_load_A_f32(const float* t, int pitch, int lane) {
  const int m  = lane & 15;
  const int g8 = (lane & 16) >> 1;
  const float* p = t + m * pitch + g8;
  v4f a0 = *(const v4f*)p;
  v4f a1 = *(const v4f*)(p + 4);
  v4f b0 = *(const v4f*)(p + 16);
  v4f b1 = *(const v4f*)(p + 20);
  v16bf r;
  #pragma unroll
  for (int i = 0; i < 4; ++i) {
    r[i]      = (__bf16)a0[i];
    r[i + 4]  = (__bf16)a1[i];
    r[i + 8]  = (__bf16)b0[i];
    r[i + 12] = (__bf16)b1[i];
  }
  return r;
}

__device__ __forceinline__ v8f wmma_bf16(v16bf a, v16bf b, v8f c) {
  return __builtin_amdgcn_wmma_f32_16x16x32_bf16(false, a, false, b, (short)0, c, false, false);
}

// Copy rows x 128 f32 (pitch 128) from global -> bf16 LDS (pitch dstPitch)
__device__ __forceinline__ void stage_f32_to_bf16(const float* __restrict__ src, __bf16* dst,
                                                  int rows, int dstPitch, int tid) {
  const int total = rows * 32;  // float4 chunks
  for (int i = tid; i < total; i += THREADS) {
    const int r = i >> 5;
    const int c = (i & 31) << 2;
    v4f v = *(const v4f*)(src + r * 128 + c);
    v4bf o;
    o[0] = (__bf16)v[0]; o[1] = (__bf16)v[1]; o[2] = (__bf16)v[2]; o[3] = (__bf16)v[3];
    *(v4bf*)(dst + r * dstPitch + c) = o;
  }
}

__global__ __launch_bounds__(THREADS, 1)
void fine_attn_kernel(const float* __restrict__ q_emb,  const float* __restrict__ kv_emb,
                      const float* __restrict__ wq_w,   const float* __restrict__ wq_b,
                      const float* __restrict__ wk_w,   const float* __restrict__ wk_b,
                      const float* __restrict__ wv_w,   const float* __restrict__ wv_b,
                      const float* __restrict__ wo_w,   const float* __restrict__ wo_b,
                      float* __restrict__ outp,         float* __restrict__ attnp) {
  extern __shared__ char smem[];
  __bf16* sKV = (__bf16*)(smem + OFF_KV);
  __bf16* sQ0 = (__bf16*)(smem + OFF_Q0);
  __bf16* sQP = (__bf16*)(smem + OFF_QP);
  __bf16* sKH = (__bf16*)(smem + OFF_KH);
  __bf16* sVT = (__bf16*)(smem + OFF_VT);
  float*  sO  = (float*)(smem + OFF_O);
  char*   sSC = smem + OFF_SC;
  float*  sMX = (float*)(smem + OFF_RED);
  float*  sIV = sMX + 32;

  const int tid  = threadIdx.x;
  const int w    = tid >> 5;
  const int lane = tid & 31;
  const int nloc = lane & 15;
  const int g    = (lane >> 4) & 1;
  const int b    = blockIdx.x;

  // ---- Phase 0: stage kv, q, wq (bf16) ----
  stage_f32_to_bf16(kv_emb + (size_t)b * LV * D, sKV, LV, KV_P, tid);
  stage_f32_to_bf16(q_emb  + (size_t)b * Bq * D, sQ0, Bq, KV_P, tid);
  stage_f32_to_bf16(wq_w, (__bf16*)sSC, D, KV_P, tid);
  __syncthreads();

  // ---- Phase 1: Q = q @ wq^T + b  (32x128), 16 tiles over 8 waves ----
  {
    const __bf16* sWq = (const __bf16*)sSC;
    for (int t = w; t < 16; t += 8) {
      const int mt = t & 1, nt = t >> 1;
      v8f acc = {};
      #pragma unroll
      for (int kc = 0; kc < 4; ++kc) {
        v16bf a  = lds_load_A (sQ0 + mt * 16 * KV_P + kc * 32, KV_P, lane);
        v16bf bb = lds_load_BT(sWq + nt * 16 * KV_P + kc * 32, KV_P, lane);
        acc = wmma_bf16(a, bb, acc);
      }
      const float bias = wq_b[nt * 16 + nloc];
      #pragma unroll
      for (int r = 0; r < 8; ++r)
        sQP[(mt * 16 + r + 8 * g) * KV_P + nt * 16 + nloc] = (__bf16)(acc[r] + bias);
    }
  }
  __syncthreads();

  const size_t attn_base_b = (size_t)b * H * Bq * LV;

  // ---- Per-head loop ----
  for (int h = 0; h < H; ++h) {
    __bf16* sWk = (__bf16*)sSC;
    __bf16* sWv = sWk + 32 * KV_P;
    stage_f32_to_bf16(wk_w + (size_t)h * 32 * 128, sWk, 32, KV_P, tid);
    stage_f32_to_bf16(wv_w + (size_t)h * 32 * 128, sWv, 32, KV_P, tid);
    __syncthreads();

    // Phase 2b: K_h (384x32, row-major) and V_h (stored transposed 32x384)
    for (int t = w; t < 96; t += 8) {
      const int buf  = t & 1;        // 0 = K, 1 = V
      const int rest = t >> 1;
      const int nt   = rest & 1;     // 0..1
      const int mt   = rest >> 1;    // 0..23
      const __bf16* sW = buf ? sWv : sWk;
      v8f acc = {};
      #pragma unroll
      for (int kc = 0; kc < 4; ++kc) {
        v16bf a  = lds_load_A (sKV + mt * 16 * KV_P + kc * 32, KV_P, lane);
        v16bf bb = lds_load_BT(sW  + nt * 16 * KV_P + kc * 32, KV_P, lane);
        acc = wmma_bf16(a, bb, acc);
      }
      const float bias = (buf ? wv_b : wk_b)[h * 32 + nt * 16 + nloc];
      if (buf == 0) {
        #pragma unroll
        for (int r = 0; r < 8; ++r)
          sKH[(mt * 16 + r + 8 * g) * KH_P + nt * 16 + nloc] = (__bf16)(acc[r] + bias);
      } else {
        #pragma unroll
        for (int r = 0; r < 8; ++r)
          sVT[(nt * 16 + nloc) * VT_P + mt * 16 + r + 8 * g] = (__bf16)(acc[r] + bias);
      }
    }
    __syncthreads();

    // Phase 2c: scores S = (Q_h K_h^T) * scale, masked.  48 tiles, 1 WMMA each (K=hd=32)
    float* sS = (float*)sSC;
    for (int t = w; t < 48; t += 8) {
      const int mt = t & 1;
      const int kt = t >> 1;  // key tile 0..23
      v16bf a  = lds_load_A (sQP + mt * 16 * KV_P + h * 32, KV_P, lane);
      v16bf bb = lds_load_BT(sKH + kt * 16 * KH_P, KH_P, lane);
      v8f acc = {};
      acc = wmma_bf16(a, bb, acc);
      const int  j      = kt * 16 + nloc;           // key index 0..383
      const bool masked = (j & 7) > (j >> 3);       // tril(L=48, V=8): keep v<=l
      #pragma unroll
      for (int r = 0; r < 8; ++r) {
        const float v = masked ? -__builtin_inff() : acc[r] * 0.17677669529663687f;
        sS[(mt * 16 + r + 8 * g) * S_P + j] = v;
      }
    }
    __syncthreads();

    // Softmax row reductions: 8 lanes per row, shfl_xor tree
    {
      const int row = tid >> 3;
      const int sub = tid & 7;
      float mx = -__builtin_inff();
      for (int j = sub; j < LV; j += 8) mx = fmaxf(mx, sS[row * S_P + j]);
      #pragma unroll
      for (int o = 1; o < 8; o <<= 1) mx = fmaxf(mx, __shfl_xor(mx, o, 32));
      float sum = 0.f;
      for (int j = sub; j < LV; j += 8) sum += __expf(sS[row * S_P + j] - mx);
      #pragma unroll
      for (int o = 1; o < 8; o <<= 1) sum += __shfl_xor(sum, o, 32);
      if (sub == 0) { sMX[row] = mx; sIV[row] = 1.0f / sum; }
    }
    __syncthreads();

    // Normalize in place + coalesced write of attn to global
    {
      float* attng = attnp + attn_base_b + (size_t)h * Bq * LV;
      for (int i = tid; i < Bq * LV; i += THREADS) {
        const int q = i / LV;
        const int j = i - q * LV;
        float* ps = sS + q * S_P + j;
        const float p = __expf(*ps - sMX[q]) * sIV[q];
        *ps = p;
        attng[(size_t)q * LV + j] = p;
      }
    }
    __syncthreads();

    // attn @ V_h -> O columns [h*32, h*32+32).  4 tiles on waves 0..3, K=384 (12 steps)
    if (w < 4) {
      const int mt = w >> 1, nt = w & 1;
      v8f acc = {};
      #pragma unroll
      for (int kc = 0; kc < 12; ++kc) {
        v16bf a  = lds_load_A_f32(sS + mt * 16 * S_P + kc * 32, S_P, lane);
        v16bf bb = lds_load_BT  (sVT + nt * 16 * VT_P + kc * 32, VT_P, lane);
        acc = wmma_bf16(a, bb, acc);
      }
      #pragma unroll
      for (int r = 0; r < 8; ++r)
        sO[(mt * 16 + r + 8 * g) * O_P + h * 32 + nt * 16 + nloc] = acc[r];
    }
    __syncthreads();
  }

  // ---- Phase 3: out = O @ wo^T + b -> global ----
  stage_f32_to_bf16(wo_w, (__bf16*)sSC, D, KV_P, tid);
  __syncthreads();
  {
    const __bf16* sWo = (const __bf16*)sSC;
    float* og = outp + (size_t)b * Bq * D;
    for (int t = w; t < 16; t += 8) {
      const int mt = t & 1, nt = t >> 1;
      v8f acc = {};
      #pragma unroll
      for (int kc = 0; kc < 4; ++kc) {
        v16bf a  = lds_load_A_f32(sO + mt * 16 * O_P + kc * 32, O_P, lane);
        v16bf bb = lds_load_BT  (sWo + nt * 16 * KV_P + kc * 32, KV_P, lane);
        acc = wmma_bf16(a, bb, acc);
      }
      const float bias = wo_b[nt * 16 + nloc];
      #pragma unroll
      for (int r = 0; r < 8; ++r)
        og[(size_t)(mt * 16 + r + 8 * g) * D + nt * 16 + nloc] = acc[r] + bias;
    }
  }
}

extern "C" void kernel_launch(void* const* d_in, const int* in_sizes, int n_in,
                              void* d_out, int out_size, void* d_ws, size_t ws_size,
                              hipStream_t stream) {
  const float* q_emb  = (const float*)d_in[0];
  const float* kv_emb = (const float*)d_in[1];
  const float* wq_w   = (const float*)d_in[2];
  const float* wq_b   = (const float*)d_in[3];
  const float* wk_w   = (const float*)d_in[4];
  const float* wk_b   = (const float*)d_in[5];
  const float* wv_w   = (const float*)d_in[6];
  const float* wv_b   = (const float*)d_in[7];
  const float* wo_w   = (const float*)d_in[8];
  const float* wo_b   = (const float*)d_in[9];

  const int Bn = in_sizes[0] / (Bq * D);   // 2048
  float* outp  = (float*)d_out;
  float* attnp = outp + (size_t)Bn * Bq * D;

  fine_attn_kernel<<<dim3(Bn), dim3(THREADS), (unsigned)SMEM_BYTES, stream>>>(
      q_emb, kv_emb, wq_w, wq_b, wk_w, wk_b, wv_w, wv_b, wo_w, wo_b, outp, attnp);
}
